// Yingram_80058190397770
// MI455X (gfx1250) — compile-verified, matrix operand order checked by reading, not run on hardware
//
#include <hip/hip_runtime.h>
#include <cmath>

typedef __attribute__((ext_vector_type(16))) _Float16 v16h;
typedef __attribute__((ext_vector_type(8)))  float    v8f;

static constexpr int FRAME_W  = 2048;        // frame length == tau_max
static constexpr int PADW     = FRAME_W + 64;
static constexpr int NTHREADS = 256;         // 8 wave32
static constexpr int NWAVES   = 8;
static constexpr int DUMP     = 480;         // dump zone for negative lags (delta >= -15)
static constexpr int RLEN     = DUMP + FRAME_W;

__global__ __launch_bounds__(NTHREADS)
void yingram_kernel(const float* __restrict__ x, float* __restrict__ out,
                    int M, float midi0)
{
    __shared__ _Float16 xh[PADW];       // frame in f16 (+ zero pad)
    __shared__ float    P[FRAME_W];     // inclusive prefix of x^2; later cumsum(d)
    __shared__ float    rD[RLEN];       // [0,480): dump; [480+k]: autocorr/d/d0
    __shared__ float    part[NTHREADS]; // block-scan partials

    const int tid   = threadIdx.x;
    const int lane  = tid & 31;
    const int wave  = tid >> 5;
    const int frame = blockIdx.x;
    const float* xg = x + (size_t)frame * FRAME_W;

    // ---- stage 1: load frame, convert to f16 in LDS, local energy prefix ----
    float4 v0 = ((const float4*)xg)[tid * 2 + 0];
    float4 v1 = ((const float4*)xg)[tid * 2 + 1];
    float xs[8] = {v0.x, v0.y, v0.z, v0.w, v1.x, v1.y, v1.z, v1.w};
    float run = 0.f;
#pragma unroll
    for (int i = 0; i < 8; ++i) {
        int idx = tid * 8 + i;
        xh[idx] = (_Float16)xs[i];
        run += xs[i] * xs[i];
        P[idx] = run;                   // thread-local inclusive, offset later
    }
    part[tid] = run;
    if (tid < PADW - FRAME_W) xh[FRAME_W + tid] = (_Float16)0.f;
#pragma unroll
    for (int i = 0; i < (RLEN + NTHREADS - 1) / NTHREADS; ++i) {
        int idx = tid + i * NTHREADS;
        if (idx < RLEN) rD[idx] = 0.f;
    }
    __syncthreads();

    // inclusive Hillis-Steele scan of the 256 partials
    for (int off = 1; off < NTHREADS; off <<= 1) {
        float v = part[tid];
        if (tid >= off) v += part[tid - off];
        __syncthreads();
        part[tid] = v;
        __syncthreads();
    }
    {
        float ofs = (tid > 0) ? part[tid - 1] : 0.f;
#pragma unroll
        for (int i = 0; i < 8; ++i) P[tid * 8 + i] += ofs;
    }
    __syncthreads();

    // ---- stage 2: autocorrelation as chunked Gram GEMMs on WMMA ----
    // Chunks of L=32: Y_s[p,i] = x[32p+s+i], p in [0,64).  C_s = Y0 * Ys^T:
    // r[32*delta + s] = sum_p C_s[p, p+delta].  Triangular pt/qt loop nest
    // (no rodata tables), A hoisted per pt.  Inner si loop (s = wave + 8*si)
    // is unrolled and software-pipelined: B for si+1 is issued before the
    // scatter of si, so the in-order DS queue lets WMMA/B-fetch overlap the
    // ds_add_f32 scatter.  Each wave: 10 tile pairs x 4 shifts = 40 WMMA.
    const int n16 = lane & 15;
    const int hi  = lane >> 4;
    for (int pt = 0; pt < 4; ++pt) {
        // A (16x32 f16): row m = chunk 16*pt+m.  ISA 16-bit A layout:
        // lanes 0-15: K = 0..7,16..23 ; lanes 16-31: K = 8..15,24..31
        const _Float16* pa = &xh[512 * pt + 32 * n16 + 8 * hi];
        v16h a;
#pragma unroll
        for (int e = 0; e < 8; ++e) { a[e] = pa[e]; a[8 + e] = pa[16 + e]; }
        for (int qt = pt; qt < 4; ++qt) {
            const int dblk = 16 * (qt - pt);
            // scatter base: k = 32*(dblk + n16 - 8*hi - v) + s + DUMP
            const int kcol = 32 * (dblk + n16 - 8 * hi) + DUMP - 224;
            float* rb0 = &rD[kcol + wave];
            // B (32x16 f16): column n = chunk 16*qt+n shifted by s (mirror
            // of the A layout)
            const _Float16* pbw = &xh[512 * qt + 32 * n16 + 8 * hi] + wave;
            v16h bcur;
#pragma unroll
            for (int e = 0; e < 8; ++e) { bcur[e] = pbw[e]; bcur[8 + e] = pbw[16 + e]; }
#pragma unroll
            for (int si = 0; si < 4; ++si) {
                v8f c = {};
                c = __builtin_amdgcn_wmma_f32_16x16x32_f16(false, a, false, bcur,
                                                           (short)0, c, false, false);
                // prefetch next shift's B before the scatter (software pipeline)
                v16h bnext = bcur;
                if (si < 3) {
                    const _Float16* pn = pbw + 8 * (si + 1);
#pragma unroll
                    for (int e = 0; e < 8; ++e) { bnext[e] = pn[e]; bnext[8 + e] = pn[16 + e]; }
                }
                // lane holds n = lane&15, m = v + 8*(lane>>4);
                // k = 32*(dblk + n - m) + s.  Unconditional adds: negative
                // deltas land in the dump zone, k==0 is overwritten later.
#pragma unroll
                for (int v = 0; v < 8; ++v)
                    atomicAdd(&rb0[(7 - v) * 32 + 8 * si], c[v]);
                bcur = bnext;
            }
        }
    }
    __syncthreads();

    // ---- stage 3: d_raw[k] = s[W-k] + s[W] - s[k] - 2 r[k], k=1..W-1 ----
    {
        const float sW = P[FRAME_W - 1];
        for (int k = tid; k < FRAME_W; k += NTHREADS)
            if (k >= 1)
                rD[DUMP + k] = P[FRAME_W - 1 - k] + sW - P[k - 1]
                               - 2.f * rD[DUMP + k];
    }
    __syncthreads();

    // ---- stage 4: inclusive cumsum of d over k=1..W-1 into P[k] ----
    {
        float run2 = 0.f;
        const int k0 = 1 + tid * 8;
#pragma unroll
        for (int i = 0; i < 8; ++i) {
            int k = k0 + i;
            if (k < FRAME_W) { run2 += rD[DUMP + k]; P[k] = run2; }
        }
        part[tid] = run2;
        __syncthreads();
        for (int off = 1; off < NTHREADS; off <<= 1) {
            float v = part[tid];
            if (tid >= off) v += part[tid - off];
            __syncthreads();
            part[tid] = v;
            __syncthreads();
        }
        float ofs2 = (tid > 0) ? part[tid - 1] : 0.f;
#pragma unroll
        for (int i = 0; i < 8; ++i) {
            int k = k0 + i;
            if (k < FRAME_W) P[k] += ofs2;
        }
    }
    __syncthreads();

    // ---- stage 5: cumulative-mean normalization; d0[0] = 1 ----
    for (int k = tid; k < FRAME_W; k += NTHREADS) {
        if (k == 0) rD[DUMP] = 1.0f;
        else        rD[DUMP + k] = (float)k * rD[DUMP + k] / (P[k] + 1e-7f);
    }
    __syncthreads();

    // ---- stage 6: linear interpolation onto midi-grid lags ----
    float* og = out + (size_t)frame * M;
    for (int i = tid; i < M; i += NTHREADS) {
        float midi = midi0 + (float)i * 0.05f;                 // 1/N_BIN
        float lag  = 22050.0f / (440.0f * exp2f((midi - 69.0f) * (1.0f / 12.0f)));
        float lf = floorf(lag), lc = ceilf(lag);
        int   fi = (int)lf,    ci = (int)lc;
        float dlo = rD[DUMP + fi];
        float y   = (ci > fi)
                  ? (lag - lf) * (rD[DUMP + ci] - dlo) / (lc - lf) + dlo
                  : dlo;
        og[i] = y;
    }
}

extern "C" void kernel_launch(void* const* d_in, const int* in_sizes, int n_in,
                              void* d_out, int out_size, void* d_ws, size_t ws_size,
                              hipStream_t stream)
{
    (void)n_in; (void)d_ws; (void)ws_size;
    const float* x = (const float*)d_in[0];
    const int frames = in_sizes[0] / FRAME_W;          // 16*1024 = 16384
    const int M      = out_size / frames;              // 1580 midi bins
    // midi_min = ceil(12*log2(sr/(440*lag_max)) + 69)  (host-side CPU math)
    const double mmin = std::ceil(12.0 * std::log2(22050.0 / (440.0 * 2048.0)) + 69.0);
    yingram_kernel<<<frames, NTHREADS, 0, stream>>>(x, (float*)d_out, M, (float)mmin);
}
